// Model_44736379355562
// MI455X (gfx1250) — matrix operand order repaired and checked by reading, not compiled, not run on hardware
//
#include <hip/hip_runtime.h>
#include <hip/hip_bf16.h>

// ---------------------------------------------------------------------------
// Gated steerable 3D conv net (4 layers, 5x5x5 kernels, pad=2) on 8x64^3.
// Implicit GEMM over the 125 taps with K = c_in padded to 32, using
// v_wmma_f32_16x16x32_f16. Each wave accumulates ZB=4 output z-planes so the
// per-tap A (weight) fragments are loaded once per 4 planes. Activations are
// fp16 [b][z][y][x][ci32]; the input window lives in an 8-plane LDS z-ring,
// staged with GLOBAL_LOAD_ASYNC_TO_LDS when the toolchain exposes it.
// ---------------------------------------------------------------------------

typedef __attribute__((ext_vector_type(16))) _Float16 v16h;
typedef __attribute__((ext_vector_type(8)))  float    v8f;

#define BATCH 8
#define DD 64
#define HH 64
#define WW 64
#define CPAD 32          // padded channel count (K dim of WMMA)
#define NB 15            // basis count
#define NTAP 125         // 5*5*5
#define XT 32            // output x per workgroup
#define TX (XT + 4)      // input x tile = 36 (halo of 2 each side)
#define NWAVE 2          // 2 waves of 32 -> 32 output x per block
#define ZB 4             // output z-planes accumulated per wave
#define RING 8           // LDS z-ring slots (ZB + 4 halo)

#define TILE_ELEMS (RING * 5 * TX * CPAD)
#define TILE_BYTES (TILE_ELEMS * 2)

// CDNA5 async global->LDS staging, if the toolchain exposes the builtins.
#if defined(__gfx1250__) && \
    __has_builtin(__builtin_amdgcn_global_load_async_to_lds_b128) && \
    __has_builtin(__builtin_amdgcn_s_wait_asynccnt)
#define USE_ASYNC 1
// parameter type per hipcc diagnostic: <4 x i32>* in AS1 (global) / AS3 (LDS)
typedef int async_v4i __attribute__((vector_size(16)));
typedef __attribute__((address_space(1))) async_v4i* async_gptr;
typedef __attribute__((address_space(3))) async_v4i* async_lptr;
#else
#define USE_ASYNC 0
#endif

// ---- per-layer compile-time configuration --------------------------------
template<int LAYER> struct Cfg;
template<> struct Cfg<1> { static constexpr int NMT=3, M0=2, NFEAT=32, NGATE=6; static constexpr bool GATED=true;  };
template<> struct Cfg<2> { static constexpr int NMT=3, M0=0, NFEAT=30, NGATE=6; static constexpr bool GATED=true;  };
template<> struct Cfg<3> { static constexpr int NMT=3, M0=2, NFEAT=32, NGATE=6; static constexpr bool GATED=true;  };
template<> struct Cfg<4> { static constexpr int NMT=1, M0=0, NFEAT=1,  NGATE=0; static constexpr bool GATED=false; };

template<int LAYER>
__device__ __forceinline__ int gate_idx(int co) {
  if (LAYER == 2) {                       // mults (0,2,2,2): feats 30
    if (co < 6)  return (co) / 3;
    if (co < 16) return 2 + (co - 6) / 5;
    return 4 + (co - 16) / 7;
  } else {                                // mults (2,2,2,2): feats 32
    if (co < 8)  return (co - 2) / 3;
    if (co < 18) return 2 + (co - 8) / 5;
    return 4 + (co - 18) / 7;
  }
}

// ---- input prep: f32 NCDHW (C=1) -> f16 [b][z][y][x][ci32] ----------------
__global__ void prep_input(const float* __restrict__ x, _Float16* __restrict__ act)
{
  size_t t = (size_t)blockIdx.x * blockDim.x + threadIdx.x;   // one 16B chunk
  size_t total = (size_t)BATCH * DD * HH * WW * (CPAD / 8);
  if (t >= total) return;
  int    ci8 = (int)(t & 3);
  size_t pos = t >> 2;
  union { _Float16 h[8]; uint4 q; } v;
  #pragma unroll
  for (int i = 0; i < 8; ++i) v.h[i] = (_Float16)0.f;
  if (ci8 == 0) v.h[0] = (_Float16)x[pos];
  *(uint4*)(act + t * 8) = v.q;
}

// ---- weight prep: K[o][i][tap] = sum_b w[o][i][b]*basis[b][tap], f16 ------
// AW[((tap*nmt + mt)*16 + m)*32 + k], zero-padded in co and ci so the WMMA
// A-fragment load is two contiguous 16B runs per lane.
__global__ void prep_weights(const float* __restrict__ w, const float* __restrict__ basis,
                             _Float16* __restrict__ AW, int c_out, int c_in, int nmt)
{
  int t = blockIdx.x * blockDim.x + threadIdx.x;
  int total = NTAP * nmt * 16 * 32;
  if (t >= total) return;
  int k   = t & 31;
  int m   = (t >> 5) & 15;
  int mt  = (t >> 9) % nmt;
  int tap = t / (512 * nmt);
  int co  = mt * 16 + m;
  float acc = 0.f;
  if (co < c_out && k < c_in) {
    #pragma unroll
    for (int bb = 0; bb < NB; ++bb)
      acc += w[(co * c_in + k) * NB + bb] * basis[bb * NTAP + tap];
  }
  AW[t] = (_Float16)acc;
}

// ---- main conv + fused gated nonlinearity ---------------------------------
template<int LAYER>
__global__ __launch_bounds__(NWAVE * 32)
void conv_wmma(const _Float16* __restrict__ actIn,
               _Float16* __restrict__ actOut,
               float* __restrict__ finalOut,
               const _Float16* __restrict__ AW)
{
  using C = Cfg<LAYER>;
  constexpr int NMT = C::NMT;

  // dynamic LDS: 8-plane z ring [slot 8][y 5][x 36][ci 32] f16 = 90 KB
  extern __shared__ __align__(16) _Float16 tile[];
  __shared__ float gateLds[ZB * NWAVE * 16 * 8];   // 4 KB static

  const int tid   = threadIdx.x;
  const int xblk  = blockIdx.x;        // 0..1
  const int y     = blockIdx.y;        // 0..63
  const int b     = blockIdx.z;        // 0..7
  const int wv    = tid >> 5;          // wave32 (CDNA5)
  const int L     = tid & 31;
  const int N     = L & 15;            // column within 16x16 tile
  const int g     = L >> 4;            // lane half
  const int xbase = xblk * XT;

  auto load_plane = [&](int zp) {
    const int slot = ((zp % RING) + RING) % RING;
    const bool zin = (unsigned)zp < (unsigned)DD;
    for (int c = tid; c < 5 * TX * (CPAD / 8); c += NWAVE * 32) {
      const int ci8 = c & 3;
      const int xx  = (c >> 2) % TX;
      const int yy  = c / (4 * TX);
      const int yg  = y - 2 + yy;
      const int xg  = xbase - 2 + xx;
      _Float16* ldst = tile + ((slot * 5 + yy) * TX + xx) * CPAD + ci8 * 8;
      if (zin && (unsigned)yg < (unsigned)HH && (unsigned)xg < (unsigned)WW) {
        const _Float16* gsrc = actIn +
            ((((size_t)b * DD + zp) * HH + yg) * WW + xg) * CPAD + ci8 * 8;
#if USE_ASYNC
        __builtin_amdgcn_global_load_async_to_lds_b128(
            (async_gptr)gsrc, (async_lptr)ldst, 0, 0);
#else
        *(uint4*)ldst = *(const uint4*)gsrc;
#endif
      } else {
        *(uint4*)ldst = make_uint4(0u, 0u, 0u, 0u);   // zero halo
      }
    }
  };

  // preload planes -2 .. 5 (zeros outside the volume)
  for (int zp = -2; zp <= ZB + 1; ++zp) load_plane(zp);

  for (int zc = 0; zc < DD; zc += ZB) {
#if USE_ASYNC
    __builtin_amdgcn_s_wait_asynccnt(0);   // async ring fills landed
#endif
    __syncthreads();                       // ring planes ready

    v8f acc[ZB][NMT];
    #pragma unroll
    for (int zz = 0; zz < ZB; ++zz)
      #pragma unroll
      for (int m = 0; m < NMT; ++m) acc[zz][m] = v8f{};

    const int xw = wv * 16 + N;            // tile-local x (halo origin -2)

    for (int kz = 0; kz < 5; ++kz) {
      int slotz[ZB];
      #pragma unroll
      for (int zz = 0; zz < ZB; ++zz)
        slotz[zz] = (zc + zz - 2 + kz + RING) & (RING - 1);
      for (int ky = 0; ky < 5; ++ky) {
        {
          // prefetch next 5-tap A slab (lane-local address)
          const int tapn = ((kz * 5 + ky) * 5 + 5) % NTAP;
          const _Float16* pf =
              AW + ((size_t)(tapn * NMT) * 16 + N) * 32 + g * 8;
          __builtin_prefetch(pf, 0, 1);
        }
        #pragma unroll
        for (int kx = 0; kx < 5; ++kx) {
          const int tap = (kz * 5 + ky) * 5 + kx;
          // A fragments: loaded ONCE per tap, reused for all ZB z-planes
          union { v16h h; uint4 q[2]; } am[NMT];
          #pragma unroll
          for (int mt = 0; mt < NMT; ++mt) {
            const _Float16* ap =
                AW + ((size_t)(tap * NMT + mt) * 16 + N) * 32 + g * 8;
            am[mt].q[0] = *(const uint4*)(ap);
            am[mt].q[1] = *(const uint4*)(ap + 16);
          }
          #pragma unroll
          for (int zz = 0; zz < ZB; ++zz) {
            // B fragment: 32(K=ci) x 16(N=x), ci contiguous in LDS
            union { v16h h; uint4 q[2]; } bm;
            const _Float16* bp =
                tile + ((slotz[zz] * 5 + ky) * TX + (xw + kx)) * CPAD + g * 8;
            bm.q[0] = *(const uint4*)(bp);
            bm.q[1] = *(const uint4*)(bp + 16);
            #pragma unroll
            for (int mt = 0; mt < NMT; ++mt) {
              acc[zz][mt] = __builtin_amdgcn_wmma_f32_16x16x32_f16(
                  false, am[mt].h, false, bm.h, (short)0, acc[zz][mt],
                  false, false);
            }
          }
        }
      }
    }

    const int xo = xbase + wv * 16 + N;

    if constexpr (C::GATED) {
      // publish sigmoid(gate) values for each column/z to LDS
      #pragma unroll
      for (int zz = 0; zz < ZB; ++zz) {
        #pragma unroll
        for (int i = 0; i < C::NGATE; ++i) {
          const int ch = C::NFEAT + i;          // constant after unroll
          const int t = ch >> 4, rr = ch & 15, gg = rr >> 3, r = rr & 7;
          if (g == gg)
            gateLds[((zz * NWAVE + wv) * 16 + N) * 8 + i] =
                1.0f / (1.0f + __expf(-acc[zz][t][r]));
        }
      }
      __syncthreads();

      #pragma unroll
      for (int zz = 0; zz < ZB; ++zz) {
        union { _Float16 h[16]; uint4 q[2]; } ob;
        #pragma unroll
        for (int t = 0; t < 2; ++t) {
          #pragma unroll
          for (int r = 0; r < 8; ++r) {
            const int co = t * 16 + g * 8 + r;  // this lane's channel
            float v = acc[zz][t][r];
            if (co < C::M0)          v = fmaxf(v, 0.f);
            else if (co < C::NFEAT)  v *= gateLds[((zz * NWAVE + wv) * 16 + N) * 8
                                                  + gate_idx<LAYER>(co)];
            else                     v = 0.f;   // gate/pad channels -> 0
            ob.h[t * 8 + r] = (_Float16)v;
          }
        }
        _Float16* op = actOut +
            ((((size_t)b * DD + (zc + zz)) * HH + y) * WW + xo) * CPAD + g * 8;
        *(uint4*)(op)      = ob.q[0];
        *(uint4*)(op + 16) = ob.q[1];
      }
    } else {
      // final layer: single fp32 output channel (lanes g==0 hold row 0)
      if (g == 0) {
        #pragma unroll
        for (int zz = 0; zz < ZB; ++zz)
          finalOut[(((size_t)b * DD + (zc + zz)) * HH + y) * WW + xo] =
              acc[zz][0][0];
      }
    }

    __syncthreads();                       // compute done; safe to reload ring
    if (zc + ZB < DD) {
      for (int zz = 0; zz < ZB; ++zz) load_plane(zc + ZB + 2 + zz);
    }
  }
}

// ---------------------------------------------------------------------------
extern "C" void kernel_launch(void* const* d_in, const int* in_sizes, int n_in,
                              void* d_out, int out_size, void* d_ws, size_t ws_size,
                              hipStream_t stream)
{
  (void)in_sizes; (void)n_in; (void)out_size; (void)ws_size;

  const float* x  = (const float*)d_in[0];
  const float* w1 = (const float*)d_in[1];
  const float* b1 = (const float*)d_in[2];
  const float* w2 = (const float*)d_in[3];
  const float* b2 = (const float*)d_in[4];
  const float* w3 = (const float*)d_in[5];
  const float* b3 = (const float*)d_in[6];
  const float* w4 = (const float*)d_in[7];
  const float* b4 = (const float*)d_in[8];

  char* ws = (char*)d_ws;
  const size_t ACT_BYTES = (size_t)BATCH * DD * HH * WW * CPAD * sizeof(_Float16);
  _Float16* actA = (_Float16*)ws;
  _Float16* actB = (_Float16*)(ws + ACT_BYTES);
  char* wp = ws + 2 * ACT_BYTES;
  const size_t AW3SZ = (size_t)NTAP * 3 * 16 * 32 * sizeof(_Float16);
  const size_t AW1SZ = (size_t)NTAP * 1 * 16 * 32 * sizeof(_Float16);
  _Float16* AW1 = (_Float16*)wp; wp += AW3SZ;
  _Float16* AW2 = (_Float16*)wp; wp += AW3SZ;
  _Float16* AW3 = (_Float16*)wp; wp += AW3SZ;
  _Float16* AW4 = (_Float16*)wp; wp += AW1SZ;

  {
    size_t total = (size_t)BATCH * DD * HH * WW * (CPAD / 8);
    prep_input<<<dim3((unsigned)((total + 255) / 256)), dim3(256), 0, stream>>>(x, actA);
  }
  auto wprep = [&](const float* w, const float* bas, _Float16* AW,
                   int co, int ci, int nmt) {
    int total = NTAP * nmt * 16 * 32;
    prep_weights<<<dim3((total + 255) / 256), dim3(256), 0, stream>>>(
        w, bas, AW, co, ci, nmt);
  };
  wprep(w1, b1, AW1, 38, 1, 3);
  wprep(w2, b2, AW2, 36, 32, 3);
  wprep(w3, b3, AW3, 38, 30, 3);
  wprep(w4, b4, AW4, 1, 32, 1);

  // allow >64KB dynamic LDS (capture-safe host-side attribute, errors ignored)
  (void)hipFuncSetAttribute((const void*)conv_wmma<1>,
      hipFuncAttributeMaxDynamicSharedMemorySize, TILE_BYTES);
  (void)hipFuncSetAttribute((const void*)conv_wmma<2>,
      hipFuncAttributeMaxDynamicSharedMemorySize, TILE_BYTES);
  (void)hipFuncSetAttribute((const void*)conv_wmma<3>,
      hipFuncAttributeMaxDynamicSharedMemorySize, TILE_BYTES);
  (void)hipFuncSetAttribute((const void*)conv_wmma<4>,
      hipFuncAttributeMaxDynamicSharedMemorySize, TILE_BYTES);

  dim3 grid(2, 64, 8), blk(NWAVE * 32);
  conv_wmma<1><<<grid, blk, TILE_BYTES, stream>>>(actA, actB, nullptr, AW1);
  conv_wmma<2><<<grid, blk, TILE_BYTES, stream>>>(actB, actA, nullptr, AW2);
  conv_wmma<3><<<grid, blk, TILE_BYTES, stream>>>(actA, actB, nullptr, AW3);
  conv_wmma<4><<<grid, blk, TILE_BYTES, stream>>>(actB, nullptr, (float*)d_out, AW4);
}
